// WindowAttention_4904852652199
// MI455X (gfx1250) — matrix operand order repaired and checked by reading, not compiled, not run on hardware
//
#include <hip/hip_runtime.h>

typedef _Float16 v16h __attribute__((ext_vector_type(16)));
typedef _Float16 v8h  __attribute__((ext_vector_type(8)));
typedef float    v8f  __attribute__((ext_vector_type(8)));

static __device__ __forceinline__ v16h join8(v8h a, v8h b) {
  return __builtin_shufflevector(a, b, 0,1,2,3,4,5,6,7,8,9,10,11,12,13,14,15);
}
static __device__ __forceinline__ v8f wmma_f16(v16h a, v16h b, v8f c) {
  return __builtin_amdgcn_wmma_f32_16x16x32_f16(false, a, false, b, (short)0, c, false, false);
}
static __device__ __forceinline__ v8h ld8h(const _Float16* p) { return *(const v8h*)p; }
static __device__ __forceinline__ v8h cvt8(const float* p) {
  float4 a = ((const float4*)p)[0];
  float4 b = ((const float4*)p)[1];
  v8h r;
  r[0] = (_Float16)a.x; r[1] = (_Float16)a.y; r[2] = (_Float16)a.z; r[3] = (_Float16)a.w;
  r[4] = (_Float16)b.x; r[5] = (_Float16)b.y; r[6] = (_Float16)b.z; r[7] = (_Float16)b.w;
  return r;
}

// ---------------- Kernel 0: bulk f32 -> f16 convert (8 elems/thread) ----------------
__global__ __launch_bounds__(256) void cvt_f32_f16_kernel(const float* __restrict__ in,
                                                          _Float16* __restrict__ out, int n8) {
  int t = blockIdx.x * 256 + threadIdx.x;
  if (t < n8) *(v8h*)(out + (size_t)t * 8) = cvt8(in + (size_t)t * 8);
}

// ---------------- Kernel 1: CPB MLP table (127 x 8) ----------------
__global__ __launch_bounds__(128) void cpb_table_kernel(
    const float* __restrict__ w1, const float* __restrict__ b1,
    const float* __restrict__ w2, float* __restrict__ table) {
  int i = threadIdx.x;
  if (i >= 127) return;
  float r = (float)(i - 63) * (8.0f / 63.0f);
  float s = (r > 0.f) ? 1.f : ((r < 0.f) ? -1.f : 0.f);
  float val = s * __log2f(fabsf(r) + 1.0f) * (1.0f / 3.0f);  // /log2(8)
  float acc[8] = {0.f, 0.f, 0.f, 0.f, 0.f, 0.f, 0.f, 0.f};
  for (int j = 0; j < 512; ++j) {
    float hj = fmaxf(val * w1[j] + b1[j], 0.f);
#pragma unroll
    for (int hh = 0; hh < 8; ++hh) acc[hh] += hj * w2[hh * 512 + j];
  }
#pragma unroll
  for (int hh = 0; hh < 8; ++hh) table[i * 8 + hh] = acc[hh];
}

// ---------------- Kernel 2: RPB expand (H,64,64) ----------------
__global__ __launch_bounds__(256) void rpb_kernel(const float* __restrict__ table,
                                                  float* __restrict__ rpb) {
  int t = blockIdx.x * 256 + threadIdx.x;  // 8*64*64 = 32768
  int h = t >> 12, ij = t & 4095, i = ij >> 6, j = ij & 63;
  float x = table[(i - j + 63) * 8 + h];
  rpb[t] = 16.0f / (1.0f + __expf(-x));
}

// ---------------- Kernel 3: QKV GEMM + scatter (128x64 tile, pipelined) ----------------
// wave w owns rows [32w, 32w+32): 2 A-frags x 4 B-frags = 8 WMMAs per K-step
__global__ __launch_bounds__(128) void qkv_kernel(
    const _Float16* __restrict__ xh, const _Float16* __restrict__ wh,
    const float* __restrict__ qb, const float* __restrict__ vb,
    _Float16* __restrict__ qbuf, _Float16* __restrict__ kbuf, _Float16* __restrict__ vbuf) {
  __shared__ _Float16 As[2][128 * 40];
  __shared__ _Float16 Bs[2][64 * 40];
  const int tid = threadIdx.x;
  const int w = tid >> 5, l = tid & 31, lm = l & 15, hi = l >> 4;
  const int mb = blockIdx.x * 128, nb = blockIdx.y * 64;
  // staging: thread covers rows rbase+{0,32,64,96} (A) and rbase+{0,32} (B), chunk ch
  const int rbase = tid >> 2, ch = tid & 3;
  const _Float16* ag = xh + (size_t)(mb + rbase) * 256 + ch * 8;
  const _Float16* bg = wh + (size_t)(nb + rbase) * 256 + ch * 8;
  const int so = rbase * 40 + ch * 8;

  v8f acc[2][4] = {};
  {  // prologue: stage K-block 0 into buffer 0
    v8h a[4], b[2];
#pragma unroll
    for (int j = 0; j < 4; ++j) a[j] = ld8h(ag + (size_t)j * 32 * 256);
#pragma unroll
    for (int j = 0; j < 2; ++j) b[j] = ld8h(bg + (size_t)j * 32 * 256);
#pragma unroll
    for (int j = 0; j < 4; ++j) *(v8h*)(&As[0][so + j * 32 * 40]) = a[j];
#pragma unroll
    for (int j = 0; j < 2; ++j) *(v8h*)(&Bs[0][so + j * 32 * 40]) = b[j];
  }
  __syncthreads();

  for (int kk = 0; kk < 8; ++kk) {
    const int cur = kk & 1;
    const bool more = (kk < 7);
    v8h a[4], b[2];
    if (more) {  // pure f16 b128 loads for next K-block, no VALU in between
      int k0 = (kk + 1) * 32;
#pragma unroll
      for (int j = 0; j < 4; ++j) a[j] = ld8h(ag + (size_t)j * 32 * 256 + k0);
#pragma unroll
      for (int j = 0; j < 2; ++j) b[j] = ld8h(bg + (size_t)j * 32 * 256 + k0);
    }
    v16h afr[2], bfr[4];
#pragma unroll
    for (int ms = 0; ms < 2; ++ms) {
      const _Float16* ap = &As[cur][(32 * w + 16 * ms + lm) * 40];
      afr[ms] = join8(ld8h(ap + hi * 16), ld8h(ap + hi * 16 + 8));
    }
#pragma unroll
    for (int nt = 0; nt < 4; ++nt) {
      const _Float16* bp = &Bs[cur][(nt * 16 + lm) * 40 + hi * 16];
      bfr[nt] = join8(ld8h(bp), ld8h(bp + 8));
    }
#pragma unroll
    for (int ms = 0; ms < 2; ++ms)
#pragma unroll
      for (int nt = 0; nt < 4; ++nt) acc[ms][nt] = wmma_f16(afr[ms], bfr[nt], acc[ms][nt]);
    if (more) {
#pragma unroll
      for (int j = 0; j < 4; ++j) *(v8h*)(&As[cur ^ 1][so + j * 32 * 40]) = a[j];
#pragma unroll
      for (int j = 0; j < 2; ++j) *(v8h*)(&Bs[cur ^ 1][so + j * 32 * 40]) = b[j];
    }
    __syncthreads();
  }
  // epilogue: bias + scatter to q/k/v (B,H,N,Dh) f16
#pragma unroll
  for (int nt = 0; nt < 4; ++nt) {
    int c = nb + nt * 16 + lm;
    int sec = c >> 8, cc = c & 255;
    float bias = (sec == 0) ? qb[cc] : ((sec == 2) ? vb[cc] : 0.0f);
    int h = cc >> 5, dh = cc & 31;
    _Float16* dst = (sec == 0) ? qbuf : ((sec == 1) ? kbuf : vbuf);
#pragma unroll
    for (int ms = 0; ms < 2; ++ms)
#pragma unroll
      for (int r = 0; r < 8; ++r) {
        int token = mb + 32 * w + 16 * ms + hi * 8 + r;
        int b2 = token >> 6, n = token & 63;
        dst[((size_t)(b2 * 8 + h) * 64 + n) * 32 + dh] = (_Float16)(acc[ms][nt][r] + bias);
      }
  }
}

// ---------------- Kernel 4: fused cosine attention per (b,h) ----------------
__global__ __launch_bounds__(128) void attn_kernel(
    const _Float16* __restrict__ qbuf, const _Float16* __restrict__ kbuf,
    const _Float16* __restrict__ vbuf, const float* __restrict__ logit_scale,
    const float* __restrict__ rpb, _Float16* __restrict__ ctx) {
  __shared__ _Float16 qs[64 * 40];
  __shared__ _Float16 ks[64 * 40];
  __shared__ _Float16 vts[32 * 72];
  __shared__ _Float16 ps[4 * 16 * 72];
  __shared__ float qninv[64];
  __shared__ float kninv[64];
  const int tid = threadIdx.x;
  const int w = tid >> 5, l = tid & 31, lm = l & 15, hi = l >> 4;
  const int bh = blockIdx.x;
  const int h = bh & 7;
  const size_t base = (size_t)bh * 64 * 32;

  // stage q, k row-major; v transposed (Dh x N)
#pragma unroll
  for (int i = 0; i < 2; ++i) {
    int c = tid * 2 + i, row = c >> 2, ch = c & 3;
    *(v8h*)(qs + row * 40 + ch * 8) = ld8h(qbuf + base + row * 32 + ch * 8);
    *(v8h*)(ks + row * 40 + ch * 8) = ld8h(kbuf + base + row * 32 + ch * 8);
    v8h vv = ld8h(vbuf + base + row * 32 + ch * 8);
#pragma unroll
    for (int u = 0; u < 8; ++u) vts[(ch * 8 + u) * 72 + row] = vv[u];
  }
  __syncthreads();
  {  // row inverse norms
    int row = tid & 63;
    const _Float16* src = (tid < 64) ? ks : qs;
    float ss = 0.f;
#pragma unroll
    for (int d = 0; d < 32; ++d) { float t = (float)src[row * 40 + d]; ss += t * t; }
    float inv = rsqrtf(fmaxf(ss, 1e-24f));
    if (tid < 64) kninv[row] = inv; else qninv[row] = inv;
  }
  __syncthreads();
  float sscale = __expf(fminf(logit_scale[h], 4.6051701859880914f));  // log(100)

  // S = q @ k^T  (wave w owns rows [16w,16w+16))
  v8f acc[4] = {};
  {
    const _Float16* ap = qs + (16 * w + lm) * 40;
    v16h a = join8(ld8h(ap + hi * 16), ld8h(ap + hi * 16 + 8));
    v16h bfr[4];
#pragma unroll
    for (int nt = 0; nt < 4; ++nt) {
      const _Float16* bp = ks + (nt * 16 + lm) * 40 + hi * 16;
      bfr[nt] = join8(ld8h(bp), ld8h(bp + 8));
    }
#pragma unroll
    for (int nt = 0; nt < 4; ++nt) acc[nt] = wmma_f16(a, bfr[nt], acc[nt]);
  }
  // fold cosine norms + logit scale, add relative position bias
  float qn[8];
#pragma unroll
  for (int r = 0; r < 8; ++r) qn[r] = qninv[16 * w + hi * 8 + r] * sscale;
#pragma unroll
  for (int nt = 0; nt < 4; ++nt) {
    float kn = kninv[nt * 16 + lm];
#pragma unroll
    for (int r = 0; r < 8; ++r) {
      int row = 16 * w + hi * 8 + r;
      float rp = rpb[((size_t)h << 12) + row * 64 + nt * 16 + lm];
      acc[nt][r] = acc[nt][r] * (qn[r] * kn) + rp;
    }
  }
  // softmax across the 64 columns (16 lanes x 4 tiles per row)
  float rinv[8];
#pragma unroll
  for (int r = 0; r < 8; ++r) {
    float m = fmaxf(fmaxf(acc[0][r], acc[1][r]), fmaxf(acc[2][r], acc[3][r]));
    m = fmaxf(m, __shfl_xor(m, 1, 32));
    m = fmaxf(m, __shfl_xor(m, 2, 32));
    m = fmaxf(m, __shfl_xor(m, 4, 32));
    m = fmaxf(m, __shfl_xor(m, 8, 32));
    float s = 0.f;
#pragma unroll
    for (int nt = 0; nt < 4; ++nt) {
      float e = __expf(acc[nt][r] - m);
      acc[nt][r] = e;
      s += e;
    }
    s += __shfl_xor(s, 1, 32);
    s += __shfl_xor(s, 2, 32);
    s += __shfl_xor(s, 4, 32);
    s += __shfl_xor(s, 8, 32);
    rinv[r] = 1.0f / s;
  }
  // re-layout P: C/D fragment -> row-major f16 in LDS
  _Float16* ph = ps + w * 16 * 72;
#pragma unroll
  for (int nt = 0; nt < 4; ++nt)
#pragma unroll
    for (int r = 0; r < 8; ++r)
      ph[(hi * 8 + r) * 72 + nt * 16 + lm] = (_Float16)acc[nt][r];
  __syncthreads();

  // O = P @ V
  v8f o[2] = {};
#pragma unroll
  for (int kb = 0; kb < 2; ++kb) {
    const _Float16* ap = ph + lm * 72 + kb * 32;
    v16h a = join8(ld8h(ap + hi * 16), ld8h(ap + hi * 16 + 8));
    v16h bfr[2];
#pragma unroll
    for (int dt = 0; dt < 2; ++dt) {
      const _Float16* bp = vts + (dt * 16 + lm) * 72 + kb * 32 + hi * 16;
      bfr[dt] = join8(ld8h(bp), ld8h(bp + 8));
    }
#pragma unroll
    for (int dt = 0; dt < 2; ++dt) o[dt] = wmma_f16(a, bfr[dt], o[dt]);
  }
  const int b = bh >> 3;
#pragma unroll
  for (int dt = 0; dt < 2; ++dt)
#pragma unroll
    for (int r = 0; r < 8; ++r) {
      int token = b * 64 + 16 * w + hi * 8 + r;
      int c = h * 32 + dt * 16 + lm;
      ctx[(size_t)token * 256 + c] = (_Float16)(o[dt][r] * rinv[r]);
    }
}

// ---------------- Kernel 5: output projection (128x64 tile, pipelined) ----------------
__global__ __launch_bounds__(128) void proj_kernel(
    const _Float16* __restrict__ ctx, const _Float16* __restrict__ wh,
    const float* __restrict__ pb, float* __restrict__ out) {
  __shared__ _Float16 As[2][128 * 40];
  __shared__ _Float16 Bs[2][64 * 40];
  const int tid = threadIdx.x;
  const int w = tid >> 5, l = tid & 31, lm = l & 15, hi = l >> 4;
  const int mb = blockIdx.x * 128, nb = blockIdx.y * 64;
  const int rbase = tid >> 2, ch = tid & 3;
  const _Float16* ag = ctx + (size_t)(mb + rbase) * 256 + ch * 8;
  const _Float16* bg = wh + (size_t)(nb + rbase) * 256 + ch * 8;
  const int so = rbase * 40 + ch * 8;

  v8f acc[2][4] = {};
  {
    v8h a[4], b[2];
#pragma unroll
    for (int j = 0; j < 4; ++j) a[j] = ld8h(ag + (size_t)j * 32 * 256);
#pragma unroll
    for (int j = 0; j < 2; ++j) b[j] = ld8h(bg + (size_t)j * 32 * 256);
#pragma unroll
    for (int j = 0; j < 4; ++j) *(v8h*)(&As[0][so + j * 32 * 40]) = a[j];
#pragma unroll
    for (int j = 0; j < 2; ++j) *(v8h*)(&Bs[0][so + j * 32 * 40]) = b[j];
  }
  __syncthreads();

  for (int kk = 0; kk < 8; ++kk) {
    const int cur = kk & 1;
    const bool more = (kk < 7);
    v8h a[4], b[2];
    if (more) {
      int k0 = (kk + 1) * 32;
#pragma unroll
      for (int j = 0; j < 4; ++j) a[j] = ld8h(ag + (size_t)j * 32 * 256 + k0);
#pragma unroll
      for (int j = 0; j < 2; ++j) b[j] = ld8h(bg + (size_t)j * 32 * 256 + k0);
    }
    v16h afr[2], bfr[4];
#pragma unroll
    for (int ms = 0; ms < 2; ++ms) {
      const _Float16* ap = &As[cur][(32 * w + 16 * ms + lm) * 40];
      afr[ms] = join8(ld8h(ap + hi * 16), ld8h(ap + hi * 16 + 8));
    }
#pragma unroll
    for (int nt = 0; nt < 4; ++nt) {
      const _Float16* bp = &Bs[cur][(nt * 16 + lm) * 40 + hi * 16];
      bfr[nt] = join8(ld8h(bp), ld8h(bp + 8));
    }
#pragma unroll
    for (int ms = 0; ms < 2; ++ms)
#pragma unroll
      for (int nt = 0; nt < 4; ++nt) acc[ms][nt] = wmma_f16(afr[ms], bfr[nt], acc[ms][nt]);
    if (more) {
#pragma unroll
      for (int j = 0; j < 4; ++j) *(v8h*)(&As[cur ^ 1][so + j * 32 * 40]) = a[j];
#pragma unroll
      for (int j = 0; j < 2; ++j) *(v8h*)(&Bs[cur ^ 1][so + j * 32 * 40]) = b[j];
    }
    __syncthreads();
  }
#pragma unroll
  for (int nt = 0; nt < 4; ++nt) {
    int c = nb + nt * 16 + lm;
    float bias = pb[c];
#pragma unroll
    for (int ms = 0; ms < 2; ++ms)
#pragma unroll
      for (int r = 0; r < 8; ++r) {
        int token = mb + 32 * w + 16 * ms + hi * 8 + r;
        out[(size_t)token * 256 + c] = acc[ms][nt][r] + bias;
      }
  }
}

extern "C" void kernel_launch(void* const* d_in, const int* in_sizes, int n_in,
                              void* d_out, int out_size, void* d_ws, size_t ws_size,
                              hipStream_t stream) {
  const float* x  = (const float*)d_in[0];
  const float* wq = (const float*)d_in[1];
  const float* qb = (const float*)d_in[2];
  const float* vb = (const float*)d_in[3];
  const float* ls = (const float*)d_in[4];
  const float* w1 = (const float*)d_in[5];
  const float* b1 = (const float*)d_in[6];
  const float* w2 = (const float*)d_in[7];
  const float* wp = (const float*)d_in[8];
  const float* pb = (const float*)d_in[9];
  float* out = (float*)d_out;

  char* ws = (char*)d_ws;
  const size_t SEG = (size_t)2048 * 8 * 64 * 32 * 2;  // 64 MiB per f16 tensor
  _Float16* qbuf = (_Float16*)(ws);
  _Float16* kbuf = (_Float16*)(ws + SEG);
  _Float16* vbuf = (_Float16*)(ws + 2 * SEG);
  _Float16* ctx  = (_Float16*)(ws + 3 * SEG);
  _Float16* xh   = (_Float16*)(ws + 4 * SEG);            // 64 MiB
  _Float16* wqh  = (_Float16*)(ws + 5 * SEG);            // 384 KiB
  _Float16* wph  = (_Float16*)(ws + 5 * SEG + (size_t)768 * 256 * 2);
  float* rpb     = (float*)(ws + 5 * SEG + (size_t)1024 * 256 * 2);
  float* table   = (float*)(ws + 5 * SEG + (size_t)1024 * 256 * 2 + (size_t)8 * 64 * 64 * 4);

  // one-time f32 -> f16 packs
  cvt_f32_f16_kernel<<<16384, 256, 0, stream>>>(x, xh, 2048 * 64 * 256 / 8);
  cvt_f32_f16_kernel<<<96, 256, 0, stream>>>(wq, wqh, 768 * 256 / 8);
  cvt_f32_f16_kernel<<<32, 256, 0, stream>>>(wp, wph, 256 * 256 / 8);

  cpb_table_kernel<<<1, 128, 0, stream>>>(w1, b1, w2, table);
  rpb_kernel<<<128, 256, 0, stream>>>(table, rpb);
  qkv_kernel<<<dim3(1024, 12), 128, 0, stream>>>(xh, wqh, qb, vb, qbuf, kbuf, vbuf);
  attn_kernel<<<16384, 128, 0, stream>>>(qbuf, kbuf, vbuf, ls, rpb, ctx);
  proj_kernel<<<dim3(1024, 4), 128, 0, stream>>>(ctx, wph, pb, out);
}
